// PRNBlock_52243982189017
// MI455X (gfx1250) — matrix operand order repaired and checked by reading, not compiled, not run on hardware
//
#include <hip/hip_runtime.h>
#include <math.h>

// ---------------- problem constants (from reference setup_inputs) ----------------
static constexpr int Bc = 4096;  // batch rows
static constexpr int Dc = 1024;  // input/output feature dim
static constexpr int Hc = 4096;  // hidden dim
static constexpr int YS = 32;    // y_size (python scalar input d_in[4], known constant)

// ---------------- tiling ----------------
// 256 threads = 8 wave32s, arranged 2 (M) x 4 (N). Each wave computes a 64x64
// C tile = 4x4 grid of 16x16 WMMA tiles (v8f accumulators, fp32).
static constexpr int BM = 128;   // block tile along M (activation rows) -> LDS via TDM
static constexpr int BN = 256;   // block tile along N (weight columns)  -> from L2

// LDS A-tile layout: row pitch = 64B data + 16B pad = 80B (40 ushorts).
// 80B pitch => row stride of 20 banks; gcd(20,64)=4 and 5 is odd, so 16
// consecutive rows' 16B chunks cover all 64 banks -> conflict-free ds_load_b128.
static constexpr int LDS_PITCH = 40;            // ushorts per row
static constexpr int LDS_TILE  = BM * LDS_PITCH; // ushorts per buffer (5120 = 10 KB)

typedef __attribute__((ext_vector_type(8)))  __bf16 v8bf;
typedef __attribute__((ext_vector_type(16))) __bf16 v16bf;
typedef __attribute__((ext_vector_type(8)))  float  v8f;

union FragU { v16bf v; v8bf h[2]; };

#define USE_TDM (__has_builtin(__builtin_amdgcn_tensor_load_to_lds) && \
                 __has_builtin(__builtin_amdgcn_s_wait_tensorcnt))

// fp32 -> bf16 round-to-nearest-even (tanh output is bounded; no NaN path needed)
__device__ __forceinline__ unsigned short f2bf_raw(float f) {
  union { float f; unsigned u; } x; x.f = f;
  unsigned r = x.u + 0x7FFFu + ((x.u >> 16) & 1u);
  return (unsigned short)(r >> 16);
}

#if USE_TDM
typedef unsigned int v4u  __attribute__((ext_vector_type(4)));
typedef int          v4i  __attribute__((ext_vector_type(4)));
typedef int          v8it __attribute__((ext_vector_type(8)));

// Issue one TDM descriptor: async load a [rows x 32] bf16 tile (row stride lda
// elements) from global into LDS at lds_byte, with 16B of LDS padding appended
// to every 64B row (pad_interval=16 DWORDs, pad_amount=4 DWORDs).
// D# layout per CDNA5 ISA §8.3 (group 0) / §8.4 (group 1). Groups 2/3 unused
// (2-D tile: tile_dim2=0); this toolchain's 6-arg builtin takes an extra
// int32x8 before cpol — zero-filled like the other disabled groups.
__device__ __forceinline__ void tdm_load_tile_async(
    const unsigned short* gsrc, int lda, int rows, unsigned lds_byte)
{
  unsigned long long ga = (unsigned long long)(size_t)gsrc;
  v4u g0 = { 0x1u,                                   // count=1 (valid), user mode
             lds_byte,                               // lds_addr
             (unsigned)ga,                           // global_addr[31:0]
             (unsigned)((ga >> 32) & 0x01FFFFFFu) |  // global_addr[56:32]
                 (2u << 30) };                       // type=2 ("image")
  v8it g1 = {
    (int)((1u << 16)      // data_size = 2 bytes
        | (1u << 20)      // pad_enable
        | (3u << 22)      // pad_interval: 2^(3+1)=16 DWORDs (one 64B row)
        | (3u << 25)),    // pad_amount : 3+1 = 4 DWORDs (16B)
    (int)(((unsigned)lda & 0xFFFFu) << 16),                  // tensor_dim0[15:0]
    (int)((((unsigned)lda >> 16) & 0xFFFFu)
        | (((unsigned)Bc & 0xFFFFu) << 16)),                 // dim0[31:16], dim1[15:0]
    (int)(32u << 16),                                        // dim1[31:16]=0, tile_dim0=32
    (int)((unsigned)rows & 0xFFFFu),                         // tile_dim1=rows, tile_dim2=0
    (int)lda,                                                // tensor_dim0_stride[31:0]
    0, 0 };
  v4i  gz4 = { 0, 0, 0, 0 };
  v8it gz8 = { 0, 0, 0, 0, 0, 0, 0, 0 };
  __builtin_amdgcn_tensor_load_to_lds(g0, g1, gz4, gz4, gz8, 0);
}
#endif

// one K-step of WMMA work on a staged LDS A-tile
__device__ __forceinline__ void compute_kstep(
    const unsigned short* __restrict__ lb,          // LDS tile base (pitch 40)
    const unsigned short* __restrict__ Bt, int ldb, int K, int kt,
    int n0, int wm, int wn, int l16, int half, v8f acc[4][4])
{
  // prefetch next K-step of the B panel (weights live in L2; warm L0)
  if (kt + 32 < K) {
#pragma unroll
    for (int f = 0; f < 4; ++f)
      __builtin_prefetch(Bt + (size_t)(n0 + wn + f * 16 + l16) * ldb +
                             (kt + 32) + half * 16, 0, 3);
  }
  // A fragments: lane l16 = row; K = half*16 .. half*16+15 contiguous
  v16bf af[4];
#pragma unroll
  for (int f = 0; f < 4; ++f) {
    int row = wm + f * 16 + l16;
    FragU fr;
    fr.h[0] = *(const v8bf*)(lb + row * LDS_PITCH + half * 16);
    fr.h[1] = *(const v8bf*)(lb + row * LDS_PITCH + half * 16 + 8);
    af[f] = fr.v;
  }
  // B fragments: lane l16 = column n; K = half*16 .. half*16+15 contiguous
  v16bf bfr[4];
#pragma unroll
  for (int f = 0; f < 4; ++f) {
    const unsigned short* bp =
        Bt + (size_t)(n0 + wn + f * 16 + l16) * ldb + kt + half * 16;
    bfr[f] = *(const v16bf*)bp;
  }
#pragma unroll
  for (int fm = 0; fm < 4; ++fm)
#pragma unroll
    for (int fn = 0; fn < 4; ++fn)
      acc[fm][fn] = __builtin_amdgcn_wmma_f32_16x16x32_bf16(
          false, af[fm], false, bfr[fn], (short)0, acc[fm][fn], false, false);
}

// Core K-loop: acc += A[m0:m0+BM, 0:K] * Bt[n0:n0+BN, 0:K]^T
// A  : bf16 row-major [*, lda] activations, staged into LDS by the TDM
// Bt : bf16 row-major [*, ldb] pre-transposed weights (Bt[n][k])
__device__ __forceinline__ void mma_tiles(
    const unsigned short* __restrict__ A,  int lda,
    const unsigned short* __restrict__ Bt, int ldb, int K,
    int m0, int n0, unsigned short* lds, v8f acc[4][4])
{
  const int tid  = threadIdx.x;
  const int lane = tid & 31;
  const int wid  = tid >> 5;
  const int wm   = (wid & 1) * 64;
  const int wn   = (wid >> 1) * 64;
  const int l16  = lane & 15;
  const int half = lane >> 4;

#if USE_TDM
  // low 32 bits of a generic LDS pointer == byte offset within the WG's LDS
  const unsigned lds_base = (unsigned)(size_t)(void*)lds;
  __syncthreads();                       // prior LDS readers (previous call) done
  if (wid == 0)
    tdm_load_tile_async(A + (size_t)m0 * lda, lda, BM, lds_base);
  int cur = 0;
  for (int kt = 0; kt < K; kt += 32) {
    __builtin_amdgcn_s_wait_tensorcnt(0);  // tile for this step landed (wave0; no-op elsewhere)
    __syncthreads();                       // tile visible to all waves; prior reads retired
    if (kt + 32 < K && wid == 0)           // overlap next DMA with this step's WMMAs
      tdm_load_tile_async(A + (size_t)m0 * lda + (kt + 32), lda, BM,
                          lds_base + (unsigned)((cur ^ 1) * LDS_TILE * 2));
    compute_kstep(lds + cur * LDS_TILE, Bt, ldb, K, kt, n0, wm, wn, l16, half, acc);
    cur ^= 1;
  }
#else
  // fallback: cooperative VGPR staging (same LDS layout, single buffer)
  for (int kt = 0; kt < K; kt += 32) {
    __syncthreads();
#pragma unroll
    for (int i = 0; i < BM / 64; ++i) {
      int e   = (tid + i * 256) * 8;
      int row = e >> 5;
      int col = e & 31;
      *(v8bf*)(lds + row * LDS_PITCH + col) =
          *(const v8bf*)(A + (size_t)(m0 + row) * lda + kt + col);
    }
    __syncthreads();
    compute_kstep(lds, Bt, ldb, K, kt, n0, wm, wn, l16, half, acc);
  }
#endif
}

// ---- step kernel 1: r = tanh(a @ W_ah), bf16 out ----
__global__ __launch_bounds__(256) void prn_step1_kernel(
    const unsigned short* __restrict__ Abf,  // [Bc, Dc]
    const unsigned short* __restrict__ WtAh, // [Hc, Dc]  (= W_ah^T, bf16)
    unsigned short* __restrict__ Rbf)        // [Bc, Hc]
{
  __shared__ __align__(16) unsigned short lds[2 * LDS_TILE];
  const int m0 = blockIdx.y * BM;
  const int n0 = blockIdx.x * BN;
  const v8f vz = {0.f,0.f,0.f,0.f,0.f,0.f,0.f,0.f};
  v8f acc[4][4];
#pragma unroll
  for (int i = 0; i < 4; ++i)
#pragma unroll
    for (int j = 0; j < 4; ++j) acc[i][j] = vz;

  mma_tiles(Abf, Dc, WtAh, Dc, Dc, m0, n0, lds, acc);

  const int lane = threadIdx.x & 31, wid = threadIdx.x >> 5;
  const int wm = (wid & 1) * 64, wn = (wid >> 1) * 64;
  const int l16 = lane & 15, half = lane >> 4;
#pragma unroll
  for (int fm = 0; fm < 4; ++fm)
#pragma unroll
    for (int fn = 0; fn < 4; ++fn)
#pragma unroll
      for (int v = 0; v < 8; ++v) {
        int m = m0 + wm + fm * 16 + half * 8 + v;  // C layout: VGPR v -> M (lane half adds 8)
        int n = n0 + wn + fn * 16 + l16;           //           lane   -> N
        Rbf[(size_t)m * Hc + n] = f2bf_raw(tanhf(acc[fm][fn][v]));
      }
}

// ---- step kernel 2: y = tanh(r @ W_ry + a @ W_xy); fp32 out + bf16 a_next ----
__global__ __launch_bounds__(256) void prn_step2_kernel(
    const unsigned short* __restrict__ Rbf,  // [Bc, Hc]
    const unsigned short* __restrict__ WtRy, // [Dc, Hc]  (= W_ry^T, bf16)
    const unsigned short* __restrict__ Abf,  // [Bc, Dc]
    const unsigned short* __restrict__ WtXy, // [Dc, Dc]  (= W_xy^T, bf16)
    float* __restrict__ Yout,                // [Bc, Dc] slice of d_out
    unsigned short* __restrict__ Anext)      // [Bc, Dc]
{
  __shared__ __align__(16) unsigned short lds[2 * LDS_TILE];
  const int m0 = blockIdx.y * BM;
  const int n0 = blockIdx.x * BN;
  const v8f vz = {0.f,0.f,0.f,0.f,0.f,0.f,0.f,0.f};
  v8f acc[4][4];
#pragma unroll
  for (int i = 0; i < 4; ++i)
#pragma unroll
    for (int j = 0; j < 4; ++j) acc[i][j] = vz;

  // fused dual reduction into the same accumulators
  mma_tiles(Rbf, Hc, WtRy, Hc, Hc, m0, n0, lds, acc);  // K = 4096
  mma_tiles(Abf, Dc, WtXy, Dc, Dc, m0, n0, lds, acc);  // K = 1024

  const int lane = threadIdx.x & 31, wid = threadIdx.x >> 5;
  const int wm = (wid & 1) * 64, wn = (wid >> 1) * 64;
  const int l16 = lane & 15, half = lane >> 4;
#pragma unroll
  for (int fm = 0; fm < 4; ++fm)
#pragma unroll
    for (int fn = 0; fn < 4; ++fn)
#pragma unroll
      for (int v = 0; v < 8; ++v) {
        int m = m0 + wm + fm * 16 + half * 8 + v;
        int n = n0 + wn + fn * 16 + l16;
        float y = tanhf(acc[fm][fn][v]);
        Yout [(size_t)m * Dc + n] = y;               // fp32 prediction (HBM-bound part)
        Anext[(size_t)m * Dc + n] = f2bf_raw(y);     // bf16 feedback for next step
      }
}

// ---- one-time per-call prep: fp32 -> bf16 transpose (weights) & convert (x) ----
__global__ __launch_bounds__(256) void transpose_to_bf16_kernel(
    const float* __restrict__ src,   // [K][N] fp32
    unsigned short* __restrict__ dst,// [N][K] bf16
    int K, int N)
{
  __shared__ float tile[32][33];
  const int n0 = blockIdx.x * 32, k0 = blockIdx.y * 32;
#pragma unroll
  for (int j = 0; j < 4; ++j)
    tile[threadIdx.y + 8 * j][threadIdx.x] =
        src[(size_t)(k0 + threadIdx.y + 8 * j) * N + n0 + threadIdx.x];
  __syncthreads();
#pragma unroll
  for (int j = 0; j < 4; ++j)
    dst[(size_t)(n0 + threadIdx.y + 8 * j) * K + k0 + threadIdx.x] =
        f2bf_raw(tile[threadIdx.x][threadIdx.y + 8 * j]);
}

__global__ __launch_bounds__(256) void f32_to_bf16_kernel(
    const float* __restrict__ s, unsigned short* __restrict__ d, int n)
{
  int i = blockIdx.x * blockDim.x + threadIdx.x;
  if (i < n) d[i] = f2bf_raw(s[i]);
}

extern "C" void kernel_launch(void* const* d_in, const int* in_sizes, int n_in,
                              void* d_out, int out_size, void* d_ws, size_t ws_size,
                              hipStream_t stream) {
  (void)in_sizes; (void)n_in; (void)out_size;
  const float* x    = (const float*)d_in[0];
  const float* W_ah = (const float*)d_in[1];
  const float* W_ry = (const float*)d_in[2];
  const float* W_xy = (const float*)d_in[3];
  // d_in[4] = y_size (== YS, compile-time constant per reference setup)
  float* out = (float*)d_out;

  // ---- workspace carve-up (66 MB total; weights+activations stay L2-resident) ----
  const size_t needed =
      ((size_t)Hc * Dc + (size_t)Dc * Hc + (size_t)Dc * Dc +
       2 * (size_t)Bc * Dc + (size_t)Bc * Hc) * sizeof(unsigned short);
  if (ws_size < needed) return;
  char* ws = (char*)d_ws;
  unsigned short* WtAh = (unsigned short*)ws; ws += (size_t)Hc * Dc * 2;
  unsigned short* WtRy = (unsigned short*)ws; ws += (size_t)Dc * Hc * 2;
  unsigned short* WtXy = (unsigned short*)ws; ws += (size_t)Dc * Dc * 2;
  unsigned short* aA   = (unsigned short*)ws; ws += (size_t)Bc * Dc * 2;
  unsigned short* aB   = (unsigned short*)ws; ws += (size_t)Bc * Dc * 2;
  unsigned short* Rbf  = (unsigned short*)ws;

  dim3 tb(32, 8);
  transpose_to_bf16_kernel<<<dim3(Hc / 32, Dc / 32), tb, 0, stream>>>(W_ah, WtAh, Dc, Hc);
  transpose_to_bf16_kernel<<<dim3(Dc / 32, Hc / 32), tb, 0, stream>>>(W_ry, WtRy, Hc, Dc);
  transpose_to_bf16_kernel<<<dim3(Dc / 32, Dc / 32), tb, 0, stream>>>(W_xy, WtXy, Dc, Dc);
  f32_to_bf16_kernel<<<(Bc * Dc) / 256, 256, 0, stream>>>(x, aA, Bc * Dc);

  unsigned short* cur = aA;
  unsigned short* nxt = aB;
  for (int s = 0; s < YS; ++s) {
    prn_step1_kernel<<<dim3(Hc / BN, Bc / BM), 256, 0, stream>>>(cur, WtAh, Rbf);
    prn_step2_kernel<<<dim3(Dc / BN, Bc / BM), 256, 0, stream>>>(
        Rbf, WtRy, cur, WtXy, out + (size_t)s * Bc * Dc, nxt);
    unsigned short* t = cur; cur = nxt; nxt = t;   // ping-pong feedback buffer
  }
}
